// MyLoss1_76235669504677
// MI455X (gfx1250) — compile-verified, matrix operand order checked by reading, not run on hardware
//
#include <hip/hip_runtime.h>
#include <hip/hip_bf16.h>
#include <stdint.h>

// ---------------------------------------------------------------------------
// MyLoss1 for MI455X (gfx1250, wave32).
//   1) sample_kernel: deterministic coprime-stride sampling + ballot/popc
//      compaction; gathers target and on-the-fly bilinear x values; zero-pads
//      sample arrays to MAXNS so downstream loads are unguarded.
//   2) pair_wmma_kernel: one wave per (batch, 16-row p-tile). Inner loop:
//      four b128 broadcast loads per operand array per 32-wide q-strip,
//      VALU abs-diff tile in f16, reduced with v_wmma_f32_16x16x32_f16
//      against an all-ones B operand (sum == matmul with ones), f32 accum.
//   3) finalize_kernel: single thread sums 504 partials in fixed order.
// ---------------------------------------------------------------------------

typedef _Float16 v16h __attribute__((ext_vector_type(16)));
typedef float    v8f  __attribute__((ext_vector_type(8)));

#define IH 240
#define IW 320
#define OH 480
#define OW 640
#define NPIX (OH * OW)      // 307200
#define NB 8
#define MAXNS 1024
#define PTILES 63           // ceil(1008/16) covers ns <= 1008

// ---------------------------------------------------------------------------
// Kernel 1: deterministic sampling + gather (bilinear computed on the fly)
// grid: (NB, 2)  block: 1024 threads (32 waves of 32)
// ---------------------------------------------------------------------------
__global__ __launch_bounds__(1024)
void sample_kernel(const float* __restrict__ input,
                   const float* __restrict__ target,
                   const uint8_t* __restrict__ my_mask,
                   const uint8_t* __restrict__ mask,
                   const int* __restrict__ snp,
                   float* __restrict__ xi, float* __restrict__ ti,
                   float* __restrict__ xo, float* __restrict__ to_)
{
    const int b  = blockIdx.x;
    const int io = blockIdx.y;            // 0 = inside, 1 = outside
    int ns = snp[0]; if (ns > MAXNS) ns = MAXNS; if (ns < 1) ns = 1;

    __shared__ int s_list[MAXNS];
    __shared__ int s_wcnt[32];
    __shared__ int s_woff[32];
    __shared__ int s_base;
    __shared__ int s_done;

    const int tid  = threadIdx.x;
    const int lane = tid & 31;
    const int wid  = tid >> 5;

    if (tid == 0) { s_base = 0; s_done = 0; }
    __syncthreads();

    const uint8_t* mm = my_mask + (size_t)b * NPIX;
    const uint8_t* vm = mask    + (size_t)b * NPIX;

    // Deterministic pseudo-random spread: stride 77773 is coprime to 307200.
    for (int j0 = 0; j0 < NPIX; j0 += 1024) {
        if (s_done) break;                 // uniform (read after barrier)
        const int j = j0 + tid;
        int idx = 0, valid = 0;
        if (j < NPIX) {
            idx = (int)(((unsigned long long)j * 77773ull) % (unsigned long long)NPIX);
            const int m0 = (mm[idx] != 0);
            const int m1 = (vm[idx] != 0);
            valid = (io == 0) ? (m0 & m1) : ((m0 ^ 1) & m1);
        }
        const unsigned bal = (unsigned)__ballot(valid);
        if (lane == 0) s_wcnt[wid] = __popc(bal);
        __syncthreads();
        if (tid == 0) {
            int acc = s_base;
            for (int w = 0; w < 32; ++w) { s_woff[w] = acc; acc += s_wcnt[w]; }
            s_base = acc;
            if (acc >= ns) s_done = 1;
        }
        __syncthreads();
        if (valid) {
            const int pos = s_woff[wid] + __popc(bal & ((1u << lane) - 1u));
            if (pos < ns) s_list[pos] = idx;
        }
        __syncthreads();
    }

    int cnt = s_base; if (cnt > ns) cnt = ns;

    const float* inp = input  + (size_t)b * IH * IW;
    const float* tgt = target + (size_t)b * NPIX;
    float* xs = ((io == 0) ? xi : xo) + (size_t)b * MAXNS;
    float* ts = ((io == 0) ? ti : to_) + (size_t)b * MAXNS;

    const float sy = (float)(IH - 1) / (float)(OH - 1);
    const float sx = (float)(IW - 1) / (float)(OW - 1);

    // Fill the full MAXNS slot: valid samples in [0,ns), zeros beyond so the
    // pair kernel can issue unguarded b128 loads over the padded tail.
    for (int s = tid; s < MAXNS; s += 1024) {
        if (s < ns) {
            const int f  = (cnt > 0) ? s_list[(s < cnt) ? s : 0] : 0;
            const int oy = f / OW;
            const int ox = f - oy * OW;
            const float fy = (float)oy * sy;
            const float fx = (float)ox * sx;
            const int y0 = (int)fy, x0 = (int)fx;
            const int y1 = min(y0 + 1, IH - 1);
            const int x1 = min(x0 + 1, IW - 1);
            const float wy = fy - (float)y0;
            const float wx = fx - (float)x0;
            const float g00 = inp[y0 * IW + x0];
            const float g01 = inp[y0 * IW + x1];
            const float g10 = inp[y1 * IW + x0];
            const float g11 = inp[y1 * IW + x1];
            xs[s] = g00 * (1.f - wy) * (1.f - wx) + g01 * (1.f - wy) * wx
                  + g10 * wy * (1.f - wx)         + g11 * wy * wx;
            ts[s] = tgt[f];
        } else {
            xs[s] = 0.f;
            ts[s] = 0.f;
        }
    }
}

// ---------------------------------------------------------------------------
// Kernel 2: pairwise tile loss, reduced via v_wmma_f32_16x16x32_f16.
// grid: (PTILES, NB)  block: 32 threads (exactly one wave; EXEC all-ones)
// ---------------------------------------------------------------------------
__global__ __launch_bounds__(32)
void pair_wmma_kernel(const float* __restrict__ xi, const float* __restrict__ ti,
                      const float* __restrict__ xo, const float* __restrict__ to_,
                      const int* __restrict__ snp,
                      float* __restrict__ partials)
{
    const int pt = blockIdx.x;
    const int b  = blockIdx.y;
    int ns = snp[0]; if (ns > MAXNS) ns = MAXNS; if (ns < 1) ns = 1;

    const int lane = threadIdx.x;          // 0..31
    const int m    = lane & 15;            // matrix row within tile
    const int hi   = lane >> 4;            // 0: lanes 0-15, 1: lanes 16-31
    const int p    = pt * 16 + m;
    const bool pok = (p < ns);

    const float* II = xi  + (size_t)b * MAXNS;
    const float* IT = ti  + (size_t)b * MAXNS;

    // p-row values (zero-padded arrays: unguarded loads are safe; pok masks).
    const float iv = pok ? II[p] : 0.f;
    const float tv = pok ? IT[p] : 0.f;

    const float4* OI4 = (const float4*)(xo  + (size_t)b * MAXNS);
    const float4* OT4 = (const float4*)(to_ + (size_t)b * MAXNS);

    // B operand = all ones: D[m][n] = sum_k A[m][k] + C[m][n]
    v16h ones;
#pragma unroll
    for (int e = 0; e < 16; ++e) ones[e] = (_Float16)1.0f;

    v8f acc = {};
    const int qstrips = (ns + 31) >> 5;
    for (int qs = 0; qs < qstrips; ++qs) {
        // 16-bit A-matrix 16x32 layout (ISA 7.12.2):
        //   lanes 0-15 hold K = {0..7, 16..23}; lanes 16-31 K = {8..15, 24..31}
        // -> this lane-half needs floats [qs*32+hi*8, +8) and [qs*32+16+hi*8, +8):
        //    four aligned float4 broadcast loads per operand array.
        const int base = qs * 8 + hi * 2;          // float4 units
        float ov[16], tvq[16];
        *(float4*)&ov[0]   = OI4[base + 0];
        *(float4*)&ov[4]   = OI4[base + 1];
        *(float4*)&ov[8]   = OI4[base + 4];
        *(float4*)&ov[12]  = OI4[base + 5];
        *(float4*)&tvq[0]  = OT4[base + 0];
        *(float4*)&tvq[4]  = OT4[base + 1];
        *(float4*)&tvq[8]  = OT4[base + 4];
        *(float4*)&tvq[12] = OT4[base + 5];

        v16h a;
#pragma unroll
        for (int e = 0; e < 16; ++e) {
            const int K = (e < 8) ? (e + hi * 8) : (e + 8 + hi * 8);
            const int q = qs * 32 + K;
            const float v = (pok && (q < ns))
                          ? fabsf(fabsf(iv - ov[e]) - fabsf(tv - tvq[e]))
                          : 0.f;
            a[e] = (_Float16)v;
        }
        acc = __builtin_amdgcn_wmma_f32_16x16x32_f16(
                  false, a, false, ones, (short)0, acc, false, false);
    }

    // All 16 columns of D are identical row-sums -> total = (sum of D)/16.
    float part = acc[0] + acc[1] + acc[2] + acc[3]
               + acc[4] + acc[5] + acc[6] + acc[7];
#pragma unroll
    for (int off = 16; off > 0; off >>= 1) part += __shfl_xor(part, off, 32);

    if (lane == 0) partials[b * PTILES + pt] = part * (1.0f / 16.0f);
}

// ---------------------------------------------------------------------------
// Kernel 3: deterministic fixed-order final reduction.
// ---------------------------------------------------------------------------
__global__ void finalize_kernel(const float* __restrict__ partials,
                                const int* __restrict__ snp,
                                float* __restrict__ out)
{
    if (threadIdx.x == 0 && blockIdx.x == 0) {
        int ns = snp[0]; if (ns > MAXNS) ns = MAXNS; if (ns < 1) ns = 1;
        float s = 0.f;
        for (int i = 0; i < NB * PTILES; ++i) s += partials[i];
        out[0] = s / ((float)ns * (float)ns);
    }
}

// ---------------------------------------------------------------------------
extern "C" void kernel_launch(void* const* d_in, const int* in_sizes, int n_in,
                              void* d_out, int out_size, void* d_ws, size_t ws_size,
                              hipStream_t stream)
{
    (void)in_sizes; (void)n_in; (void)out_size; (void)ws_size;

    const float*   input   = (const float*)d_in[0];   // (8,1,240,320) f32
    const float*   target  = (const float*)d_in[1];   // (8,1,480,640) f32
    const uint8_t* my_mask = (const uint8_t*)d_in[2]; // (8,1,480,640) bool
    const uint8_t* mask    = (const uint8_t*)d_in[3]; // (8,1,480,640) bool
    const int*     snp     = (const int*)d_in[4];     // scalar sample_num

    float* ws = (float*)d_ws;
    float* xi       = ws;                       // NB*MAXNS
    float* ti       = xi  + NB * MAXNS;         // NB*MAXNS
    float* xo       = ti  + NB * MAXNS;         // NB*MAXNS
    float* to_      = xo  + NB * MAXNS;         // NB*MAXNS
    float* partials = to_ + NB * MAXNS;         // NB*PTILES

    dim3 g1(NB, 2);
    sample_kernel<<<g1, 1024, 0, stream>>>(input, target, my_mask, mask, snp,
                                           xi, ti, xo, to_);

    dim3 g2(PTILES, NB);
    pair_wmma_kernel<<<g2, 32, 0, stream>>>(xi, ti, xo, to_, snp, partials);

    finalize_kernel<<<1, 32, 0, stream>>>(partials, snp, (float*)d_out);
}
